// AttentionModule_57475252355509
// MI455X (gfx1250) — compile-verified, matrix operand order checked by reading, not run on hardware
//
#include <hip/hip_runtime.h>
#include <math.h>

typedef _Float16 h16;
typedef h16  v8h  __attribute__((ext_vector_type(8)));
typedef h16  v16h __attribute__((ext_vector_type(16)));
typedef float v8f __attribute__((ext_vector_type(8)));

#define DIMC   512
#define HEADS  8
#define HDIM   64
#define MLPH   2048
#define BATCH  32
#define SEQ    256
#define MTOK   (BATCH * SEQ)          // 8192 tokens
#define ZBH    (BATCH * HEADS)        // 256 attention problems
#define ATT_SCALE 0.125f              // 64^-0.5

// ---------------------------------------------------------------------------
// LayerNorm: fp32 [rows, 512] -> f16 [rows, 512].  One block per row.
// ---------------------------------------------------------------------------
__global__ __launch_bounds__(256) void ln_kernel(const float* __restrict__ x,
                                                 const float* __restrict__ g,
                                                 const float* __restrict__ b,
                                                 h16* __restrict__ out) {
  const int C = DIMC;
  long long row = blockIdx.x;
  const float* xr = x + row * C;
  int tid = threadIdx.x;
  float v0 = xr[tid], v1 = xr[tid + 256];
  float s = v0 + v1, s2 = v0 * v0 + v1 * v1;
#pragma unroll
  for (int off = 16; off > 0; off >>= 1) {
    s  += __shfl_xor(s, off, 32);
    s2 += __shfl_xor(s2, off, 32);
  }
  __shared__ float red[18];
  int wave = tid >> 5, lane = tid & 31;
  if (lane == 0) { red[wave] = s; red[8 + wave] = s2; }
  __syncthreads();
  if (tid == 0) {
    float ts = 0.f, ts2 = 0.f;
#pragma unroll
    for (int i = 0; i < 8; ++i) { ts += red[i]; ts2 += red[8 + i]; }
    float m = ts / (float)C;
    float var = ts2 / (float)C - m * m;
    red[16] = m;
    red[17] = rsqrtf(var + 1e-5f);
  }
  __syncthreads();
  float m = red[16], rs = red[17];
  out[row * C + tid]       = (h16)((v0 - m) * rs * g[tid] + b[tid]);
  out[row * C + tid + 256] = (h16)((v1 - m) * rs * g[tid + 256] + b[tid + 256]);
}

// ---------------------------------------------------------------------------
// Weight convert: W[K,N] fp32 (row-major) -> WT[N,K] f16 (row-major)
// ---------------------------------------------------------------------------
__global__ __launch_bounds__(256) void wconv_kernel(const float* __restrict__ W,
                                                    h16* __restrict__ WT,
                                                    int K, int N) {
  long long t = (long long)blockIdx.x * 256 + threadIdx.x;
  if (t >= (long long)K * N) return;
  int k = (int)(t % K);
  int n = (int)(t / K);
  WT[t] = (h16)W[(long long)k * N + n];
}

// ---------------------------------------------------------------------------
// V repack: vt[z=(b*8+h)][d][n] = qkv_f16[(b*256+n)*W + voff + h*64 + d]
// ---------------------------------------------------------------------------
__global__ __launch_bounds__(256) void vtrans_kernel(const h16* __restrict__ src,
                                                     int W, int voff,
                                                     h16* __restrict__ vt) {
  long long t = (long long)blockIdx.x * 256 + threadIdx.x; // < ZBH*HDIM*SEQ
  int n = (int)(t & 255);
  long long t2 = t >> 8;
  int d = (int)(t2 & 63);
  int z = (int)(t2 >> 6);
  int bb = z >> 3, hh = z & 7;
  vt[t] = src[((long long)(bb * SEQ + n)) * W + voff + hh * HDIM + d];
}

// ---------------------------------------------------------------------------
// Row softmax over f16 [rows, 256], in place. One wave per row (wave32).
// ---------------------------------------------------------------------------
__global__ __launch_bounds__(256) void softmax_kernel(h16* __restrict__ s) {
  long long row = (long long)blockIdx.x * 8 + (threadIdx.x >> 5);
  int lane = threadIdx.x & 31;
  h16* p = s + row * 256 + lane * 8;
  v8h vv = *(v8h*)p;
  float f[8];
  float mx = -1e30f;
#pragma unroll
  for (int i = 0; i < 8; ++i) { f[i] = (float)vv[i]; mx = fmaxf(mx, f[i]); }
#pragma unroll
  for (int off = 16; off > 0; off >>= 1) mx = fmaxf(mx, __shfl_xor(mx, off, 32));
  float sum = 0.f;
#pragma unroll
  for (int i = 0; i < 8; ++i) { f[i] = __expf(f[i] - mx); sum += f[i]; }
#pragma unroll
  for (int off = 16; off > 0; off >>= 1) sum += __shfl_xor(sum, off, 32);
  float inv = 1.0f / sum;
#pragma unroll
  for (int i = 0; i < 8; ++i) vv[i] = (h16)(f[i] * inv);
  *(v8h*)p = vv;
}

// ---------------------------------------------------------------------------
// Batched WMMA GEMM:  C[z] = A[z] (MxK, lda) * BT[z]^T  (BT is NxK, ldbt)
// Block tile 256x64, BK=32 (one v_wmma_f32_16x16x32_f16 K-step).
// 8 waves stacked along M; each wave computes a 2x4 grid of 16x16 tiles
// (32 rows x 64 cols) -> 8 WMMA per K-step per wave.
// Global->LDS stage is register double-buffered: next K-tile's
// global_load_b128s are issued before the current tile's WMMAs, hiding
// HBM/L2 latency behind matrix math.
// Batch offset: z -> (zb = z/ZH, zh = z%ZH), off = zb*s?b + zh*s?h.
// Epilogue: x = acc*scale (+bias[n]) (gelu?) ; Cf[idx]=x(+resid[idx]); Ch[idx]=f16(x)
// ---------------------------------------------------------------------------
__global__ __launch_bounds__(256) void gemm_bt_kernel(
    const h16* __restrict__ A, long long sAb, long long sAh, int lda,
    const h16* __restrict__ BT, long long sBb, long long sBh, int ldbt,
    float* __restrict__ Cf, h16* __restrict__ Ch,
    long long sCb, long long sCh, int ldc,
    const float* __restrict__ bias, const float* __restrict__ resid,
    int K, int ZH, float scale, int act) {
  __shared__ h16 lA[256 * 32];   // 16 KB
  __shared__ h16 lB[64 * 32];    //  4 KB

  const int z  = blockIdx.z;
  const int zb = z / ZH, zh = z - zb * ZH;
  const h16* Ab = A  + zb * sAb + zh * sAh;
  const h16* Bb = BT + zb * sBb + zh * sBh;
  const long long offC = zb * sCb + zh * sCh;

  const int tid  = threadIdx.x;
  const int wave = tid >> 5, lane = tid & 31;
  const int m0 = wave * 32;     // wave tile origin (M) within block tile
  const long long blockM = (long long)blockIdx.x * 256;
  const long long blockN = (long long)blockIdx.y * 64;

  const int r  = lane & 15;     // fragment row (M or N index)
  const int kb = (lane >> 4) * 8;

  v8f acc[2][4] = {};

  // Global staging: each thread owns one full A row (32 halves = 4 x b128)
  // and one 8-half B chunk.
  const int brow = tid >> 2, bkk = (tid & 3) * 8;
  const h16* aSrc = Ab + (blockM + tid) * lda;
  const h16* bSrc = Bb + (blockN + brow) * ldbt + bkk;

  v8h aReg[4], bReg;
#pragma unroll
  for (int j = 0; j < 4; ++j) aReg[j] = *(const v8h*)(aSrc + 8 * j);
  bReg = *(const v8h*)(bSrc);
#pragma unroll
  for (int j = 0; j < 4; ++j) *(v8h*)&lA[tid * 32 + 8 * j] = aReg[j];
  *(v8h*)&lB[brow * 32 + bkk] = bReg;
  __syncthreads();

  for (int k0 = 0; k0 < K; k0 += 32) {
    const bool hasNext = (k0 + 32) < K;
    if (hasNext) {  // issue next tile's global loads; consumed after barrier
#pragma unroll
      for (int j = 0; j < 4; ++j)
        aReg[j] = *(const v8h*)(aSrc + k0 + 32 + 8 * j);
      bReg = *(const v8h*)(bSrc + k0 + 32);
      if (k0 + 64 < K) {
        __builtin_prefetch((const void*)(aSrc + k0 + 64), 0, 0);
        __builtin_prefetch((const void*)(bSrc + k0 + 64), 0, 0);
      }
    }

    v16h af[2], bf[4];
#pragma unroll
    for (int mt = 0; mt < 2; ++mt) {
      const h16* ptr = &lA[(m0 + mt * 16 + r) * 32 + kb];
      v8h lo = *(const v8h*)ptr;
      v8h hi = *(const v8h*)(ptr + 16);
      af[mt] = __builtin_shufflevector(lo, hi, 0, 1, 2, 3, 4, 5, 6, 7,
                                       8, 9, 10, 11, 12, 13, 14, 15);
    }
#pragma unroll
    for (int nt = 0; nt < 4; ++nt) {
      const h16* ptr = &lB[(nt * 16 + r) * 32 + kb];
      v8h lo = *(const v8h*)ptr;
      v8h hi = *(const v8h*)(ptr + 16);
      bf[nt] = __builtin_shufflevector(lo, hi, 0, 1, 2, 3, 4, 5, 6, 7,
                                       8, 9, 10, 11, 12, 13, 14, 15);
    }
#pragma unroll
    for (int mt = 0; mt < 2; ++mt)
#pragma unroll
      for (int nt = 0; nt < 4; ++nt)
        acc[mt][nt] = __builtin_amdgcn_wmma_f32_16x16x32_f16(
            false, af[mt], false, bf[nt], (short)0, acc[mt][nt], false, false);
    __syncthreads();   // all LDS reads of this tile done

    if (hasNext) {
#pragma unroll
      for (int j = 0; j < 4; ++j) *(v8h*)&lA[tid * 32 + 8 * j] = aReg[j];
      *(v8h*)&lB[brow * 32 + bkk] = bReg;
      __syncthreads(); // next tile visible to all waves
    }
  }

  const int col   = lane & 15;
  const int rbase = (lane >> 4) * 8;
#pragma unroll
  for (int mt = 0; mt < 2; ++mt)
#pragma unroll
    for (int nt = 0; nt < 4; ++nt)
#pragma unroll
      for (int v = 0; v < 8; ++v) {
        long long gM = blockM + m0 + mt * 16 + rbase + v;
        long long gN = blockN + nt * 16 + col;
        float x = acc[mt][nt][v] * scale;
        if (bias) x += bias[gN];
        if (act)  x = 0.5f * x * (1.0f + erff(x * 0.70710678118654752f));
        long long idx = offC + gM * (long long)ldc + gN;
        if (Cf) Cf[idx] = resid ? (x + resid[idx]) : x;
        if (Ch) Ch[idx] = (h16)x;
      }
}

// ---------------------------------------------------------------------------
// Host-side helpers
// ---------------------------------------------------------------------------
static void gemm(hipStream_t st,
                 const h16* A, long long sAb, long long sAh, int lda,
                 const h16* BT, long long sBb, long long sBh, int ldbt,
                 float* Cf, h16* Ch, long long sCb, long long sCh, int ldc,
                 const float* bias, const float* resid,
                 int M, int N, int K, int Z, int ZH, float scale, int act) {
  dim3 grid(M / 256, N / 64, Z);
  gemm_bt_kernel<<<grid, 256, 0, st>>>(A, sAb, sAh, lda, BT, sBb, sBh, ldbt,
                                       Cf, Ch, sCb, sCh, ldc, bias, resid,
                                       K, ZH, scale, act);
}

static void conv_w(hipStream_t st, const float* W, h16* WT, int K, int N) {
  long long tot = (long long)K * N;
  wconv_kernel<<<(int)((tot + 255) / 256), 256, 0, st>>>(W, WT, K, N);
}

struct Scratch {
  h16 *lnbuf, *lnbuf2, *qkvbuf, *qbuf, *sbuf, *vtbuf, *attnbuf, *hidbuf;
};

// x = x + attn(LN(x)); x = x + mlp(LN(x)).  X fp32 input, OUT fp32 (residual store).
static void run_self_block(hipStream_t st, const float* X, float* OUT,
                           const float* n1g, const float* n1b, const h16* qkvT,
                           const h16* projT, const float* projb,
                           const float* n2g, const float* n2b,
                           const h16* fc1T, const float* fc1b,
                           const h16* fc2T, const float* fc2b, const Scratch& w) {
  ln_kernel<<<MTOK, 256, 0, st>>>(X, n1g, n1b, w.lnbuf);
  // qkv = ln @ qkv_w  -> f16 [8192, 1536]
  gemm(st, w.lnbuf, 0, 0, DIMC, qkvT, 0, 0, DIMC,
       nullptr, w.qkvbuf, 0, 0, 3 * DIMC, nullptr, nullptr,
       MTOK, 3 * DIMC, DIMC, 1, 1, 1.f, 0);
  // S[z] = Q K^T * scale (q cols 0..511, k cols 512..1023 of qkv)
  gemm(st, w.qkvbuf, (long long)SEQ * 3 * DIMC, HDIM, 3 * DIMC,
       w.qkvbuf + DIMC, (long long)SEQ * 3 * DIMC, HDIM, 3 * DIMC,
       nullptr, w.sbuf, (long long)HEADS * SEQ * SEQ, (long long)SEQ * SEQ, SEQ,
       nullptr, nullptr, SEQ, SEQ, HDIM, ZBH, HEADS, ATT_SCALE, 0);
  softmax_kernel<<<(ZBH * SEQ) / 8, 256, 0, st>>>(w.sbuf);
  vtrans_kernel<<<(ZBH * HDIM * SEQ) / 256, 256, 0, st>>>(w.qkvbuf, 3 * DIMC,
                                                          2 * DIMC, w.vtbuf);
  // O[z] = P V  -> attnbuf f16 [8192, 512] with heads merged
  gemm(st, w.sbuf, (long long)HEADS * SEQ * SEQ, (long long)SEQ * SEQ, SEQ,
       w.vtbuf, (long long)HEADS * HDIM * SEQ, (long long)HDIM * SEQ, SEQ,
       nullptr, w.attnbuf, (long long)SEQ * DIMC, HDIM, DIMC,
       nullptr, nullptr, SEQ, HDIM, SEQ, ZBH, HEADS, 1.f, 0);
  // proj + bias + residual -> fp32 OUT
  gemm(st, w.attnbuf, 0, 0, DIMC, projT, 0, 0, DIMC,
       OUT, nullptr, 0, 0, DIMC, projb, X, MTOK, DIMC, DIMC, 1, 1, 1.f, 0);
  // MLP
  ln_kernel<<<MTOK, 256, 0, st>>>(OUT, n2g, n2b, w.lnbuf);
  gemm(st, w.lnbuf, 0, 0, DIMC, fc1T, 0, 0, DIMC,
       nullptr, w.hidbuf, 0, 0, MLPH, fc1b, nullptr,
       MTOK, MLPH, DIMC, 1, 1, 1.f, 1);
  gemm(st, w.hidbuf, 0, 0, MLPH, fc2T, 0, 0, MLPH,
       OUT, nullptr, 0, 0, DIMC, fc2b, OUT, MTOK, DIMC, MLPH, 1, 1, 1.f, 0);
}

// q = q + cross_attn(LN(q), LN(kv)); q = q + mlp(LN(q)).  QOUT fp32 in/out.
static void run_cross_block(hipStream_t st, float* QOUT, const float* KVX,
                            const float* nkvg, const float* nkvb,
                            const float* n1g, const float* n1b,
                            const h16* qT, const h16* kvT,
                            const h16* projT, const float* projb,
                            const float* n2g, const float* n2b,
                            const h16* fc1T, const float* fc1b,
                            const h16* fc2T, const float* fc2b, const Scratch& w) {
  ln_kernel<<<MTOK, 256, 0, st>>>(KVX, nkvg, nkvb, w.lnbuf2);
  ln_kernel<<<MTOK, 256, 0, st>>>(QOUT, n1g, n1b, w.lnbuf);
  gemm(st, w.lnbuf, 0, 0, DIMC, qT, 0, 0, DIMC,
       nullptr, w.qbuf, 0, 0, DIMC, nullptr, nullptr,
       MTOK, DIMC, DIMC, 1, 1, 1.f, 0);
  gemm(st, w.lnbuf2, 0, 0, DIMC, kvT, 0, 0, DIMC,
       nullptr, w.qkvbuf, 0, 0, 2 * DIMC, nullptr, nullptr,
       MTOK, 2 * DIMC, DIMC, 1, 1, 1.f, 0);
  gemm(st, w.qbuf, (long long)SEQ * DIMC, HDIM, DIMC,
       w.qkvbuf, (long long)SEQ * 2 * DIMC, HDIM, 2 * DIMC,
       nullptr, w.sbuf, (long long)HEADS * SEQ * SEQ, (long long)SEQ * SEQ, SEQ,
       nullptr, nullptr, SEQ, SEQ, HDIM, ZBH, HEADS, ATT_SCALE, 0);
  softmax_kernel<<<(ZBH * SEQ) / 8, 256, 0, st>>>(w.sbuf);
  vtrans_kernel<<<(ZBH * HDIM * SEQ) / 256, 256, 0, st>>>(w.qkvbuf, 2 * DIMC,
                                                          DIMC, w.vtbuf);
  gemm(st, w.sbuf, (long long)HEADS * SEQ * SEQ, (long long)SEQ * SEQ, SEQ,
       w.vtbuf, (long long)HEADS * HDIM * SEQ, (long long)HDIM * SEQ, SEQ,
       nullptr, w.attnbuf, (long long)SEQ * DIMC, HDIM, DIMC,
       nullptr, nullptr, SEQ, HDIM, SEQ, ZBH, HEADS, 1.f, 0);
  gemm(st, w.attnbuf, 0, 0, DIMC, projT, 0, 0, DIMC,
       QOUT, nullptr, 0, 0, DIMC, projb, QOUT, MTOK, DIMC, DIMC, 1, 1, 1.f, 0);
  ln_kernel<<<MTOK, 256, 0, st>>>(QOUT, n2g, n2b, w.lnbuf);
  gemm(st, w.lnbuf, 0, 0, DIMC, fc1T, 0, 0, DIMC,
       nullptr, w.hidbuf, 0, 0, MLPH, fc1b, nullptr,
       MTOK, MLPH, DIMC, 1, 1, 1.f, 1);
  gemm(st, w.hidbuf, 0, 0, MLPH, fc2T, 0, 0, MLPH,
       QOUT, nullptr, 0, 0, DIMC, fc2b, QOUT, MTOK, DIMC, MLPH, 1, 1, 1.f, 0);
}

// ---------------------------------------------------------------------------
// Input flattening order (setup_inputs dict order, depth-first):
// 0:h_feat 1:l_feat | h_self@2 l_self@13 lh_cross@24 hl_cross@38
// common: +0 n1_g +1 n1_b +2 proj_w +3 proj_b +4 n2_g +5 n2_b
//         +6 fc1_w +7 fc1_b +8 fc2_w +9 fc2_b
// self: +10 qkv_w ; cross: +10 q_w +11 kv_w +12 nkv_g +13 nkv_b
// ---------------------------------------------------------------------------
extern "C" void kernel_launch(void* const* d_in, const int* in_sizes, int n_in,
                              void* d_out, int out_size, void* d_ws, size_t ws_size,
                              hipStream_t stream) {
  (void)in_sizes; (void)n_in; (void)out_size; (void)ws_size;
  const float* h_feat = (const float*)d_in[0];
  const float* l_feat = (const float*)d_in[1];
  auto F = [&](int i) { return (const float*)d_in[i]; };
  const int HS = 2, LS = 13, LH = 24, HL = 38;

  // bump allocator over workspace (256B aligned)
  char* p = (char*)d_ws;
  auto ah = [&](long long nHalves) -> h16* {
    h16* r = (h16*)p;
    p += ((nHalves * 2 + 255) / 256) * 256;
    return r;
  };

  // f16 transposed weights
  h16* hs_qkvT = ah((long long)DIMC * 3 * DIMC);
  h16* hs_projT = ah((long long)DIMC * DIMC);
  h16* hs_fc1T = ah((long long)DIMC * MLPH);
  h16* hs_fc2T = ah((long long)MLPH * DIMC);
  h16* ls_qkvT = ah((long long)DIMC * 3 * DIMC);
  h16* ls_projT = ah((long long)DIMC * DIMC);
  h16* ls_fc1T = ah((long long)DIMC * MLPH);
  h16* ls_fc2T = ah((long long)MLPH * DIMC);
  h16* lh_qT = ah((long long)DIMC * DIMC);
  h16* lh_kvT = ah((long long)DIMC * 2 * DIMC);
  h16* lh_projT = ah((long long)DIMC * DIMC);
  h16* lh_fc1T = ah((long long)DIMC * MLPH);
  h16* lh_fc2T = ah((long long)MLPH * DIMC);
  h16* hl_qT = ah((long long)DIMC * DIMC);
  h16* hl_kvT = ah((long long)DIMC * 2 * DIMC);
  h16* hl_projT = ah((long long)DIMC * DIMC);
  h16* hl_fc1T = ah((long long)DIMC * MLPH);
  h16* hl_fc2T = ah((long long)MLPH * DIMC);

  Scratch w;
  w.lnbuf   = ah((long long)MTOK * DIMC);
  w.lnbuf2  = ah((long long)MTOK * DIMC);
  w.qkvbuf  = ah((long long)MTOK * 3 * DIMC);
  w.qbuf    = ah((long long)MTOK * DIMC);
  w.sbuf    = ah((long long)ZBH * SEQ * SEQ);
  w.vtbuf   = ah((long long)ZBH * HDIM * SEQ);
  w.attnbuf = ah((long long)MTOK * DIMC);
  w.hidbuf  = ah((long long)MTOK * MLPH);

  conv_w(stream, F(HS + 10), hs_qkvT, DIMC, 3 * DIMC);
  conv_w(stream, F(HS + 2),  hs_projT, DIMC, DIMC);
  conv_w(stream, F(HS + 6),  hs_fc1T, DIMC, MLPH);
  conv_w(stream, F(HS + 8),  hs_fc2T, MLPH, DIMC);
  conv_w(stream, F(LS + 10), ls_qkvT, DIMC, 3 * DIMC);
  conv_w(stream, F(LS + 2),  ls_projT, DIMC, DIMC);
  conv_w(stream, F(LS + 6),  ls_fc1T, DIMC, MLPH);
  conv_w(stream, F(LS + 8),  ls_fc2T, MLPH, DIMC);
  conv_w(stream, F(LH + 10), lh_qT, DIMC, DIMC);
  conv_w(stream, F(LH + 11), lh_kvT, DIMC, 2 * DIMC);
  conv_w(stream, F(LH + 2),  lh_projT, DIMC, DIMC);
  conv_w(stream, F(LH + 6),  lh_fc1T, DIMC, MLPH);
  conv_w(stream, F(LH + 8),  lh_fc2T, MLPH, DIMC);
  conv_w(stream, F(HL + 10), hl_qT, DIMC, DIMC);
  conv_w(stream, F(HL + 11), hl_kvT, DIMC, 2 * DIMC);
  conv_w(stream, F(HL + 2),  hl_projT, DIMC, DIMC);
  conv_w(stream, F(HL + 6),  hl_fc1T, DIMC, MLPH);
  conv_w(stream, F(HL + 8),  hl_fc2T, MLPH, DIMC);

  float* OUT_H = (float*)d_out;                         // [32,256,512]
  float* OUT_L = (float*)d_out + (long long)MTOK * DIMC;

  // h = self_block(h_feat); l = self_block(l_feat)
  run_self_block(stream, h_feat, OUT_H, F(HS + 0), F(HS + 1), hs_qkvT,
                 hs_projT, F(HS + 3), F(HS + 4), F(HS + 5),
                 hs_fc1T, F(HS + 7), hs_fc2T, F(HS + 9), w);
  run_self_block(stream, l_feat, OUT_L, F(LS + 0), F(LS + 1), ls_qkvT,
                 ls_projT, F(LS + 3), F(LS + 4), F(LS + 5),
                 ls_fc1T, F(LS + 7), ls_fc2T, F(LS + 9), w);
  // l = cross_block(l, h);  h = cross_block(h, l_new)
  run_cross_block(stream, OUT_L, OUT_H, F(LH + 12), F(LH + 13),
                  F(LH + 0), F(LH + 1), lh_qT, lh_kvT, lh_projT, F(LH + 3),
                  F(LH + 4), F(LH + 5), lh_fc1T, F(LH + 7), lh_fc2T, F(LH + 9), w);
  run_cross_block(stream, OUT_H, OUT_L, F(HL + 12), F(HL + 13),
                  F(HL + 0), F(HL + 1), hl_qT, hl_kvT, hl_projT, F(HL + 3),
                  F(HL + 4), F(HL + 5), hl_fc1T, F(HL + 7), hl_fc2T, F(HL + 9), w);
}
